// QuantLinear_16192026706132
// MI455X (gfx1250) — matrix operand order repaired and checked by reading, not compiled, not run on hardware
//
#include <hip/hip_runtime.h>

// ---- CDNA5 WMMA / vector types ---------------------------------------------
typedef __attribute__((ext_vector_type(16))) __bf16 v16bf;
typedef __attribute__((ext_vector_type(8)))  __bf16 v8bf;
typedef __attribute__((ext_vector_type(8)))  float  v8f;
typedef __attribute__((ext_vector_type(4)))  unsigned int v4u;
typedef __attribute__((ext_vector_type(8)))  int v8i_t;
typedef __attribute__((ext_vector_type(4)))  int v4i_t;

#define IN_F   4096
#define OUT_F  4096
#define GROUPQ 128
#define BM     128
#define BN     128
#define BK     32
#define LDT    40                   // padded LDS stride (bf16) per 32-elem row
#define TILE_ELEMS (128 * 32)       // 4096 bf16 = 8KB per global tile
#define TILE_LDS   (128 * LDT)      // 5120 bf16 = 10240B per LDS tile (padded)
#define NKT    (IN_F / BK)          // 128 K-steps

#ifndef HAVE_TDM
#if defined(__has_builtin)
#if __has_builtin(__builtin_amdgcn_tensor_load_to_lds) && \
    __has_builtin(__builtin_amdgcn_s_wait_tensorcnt)
#define HAVE_TDM 1
#endif
#endif
#endif
#ifndef HAVE_TDM
#define HAVE_TDM 0
#endif

union V16U { v16bf v; v8bf h[2]; };

// ============================================================================
// TDM helper: 1-descriptor contiguous 8KB tile -> LDS, with LDS padding of
// 4 dwords every 16 dwords (=> 40-bf16 row stride in LDS).
// ============================================================================
#if HAVE_TDM
__device__ __forceinline__ void tdm_tile_to_lds(const __bf16* gsrc, unsigned lds_off) {
    unsigned long long ga = (unsigned long long)(size_t)gsrc;
    v4u g0 = { 1u,                                   // count=1, user mode
               lds_off,                              // LDS byte address
               (unsigned)ga,                         // global_addr[31:0]
               ((unsigned)(ga >> 32) & 0x01FFFFFFu) | (2u << 30) }; // [56:32]|type=2
    v8i_t g1 = { (int)((1u << 16) | (1u << 20) | (3u << 22) | (3u << 25)),
                 (int)0x10000000u,   // tensor_dim0[15:0]=4096 in dw1[31:16]
                 (int)0x00010000u,   // tensor_dim0[31:16]=0, tensor_dim1[15:0]=1
                 (int)0x10000000u,   // tensor_dim1[31:16]=0, tile_dim0=4096
                 1,                  // tile_dim1=1, tile_dim2=0
                 4096,               // tensor_dim0_stride[31:0]
                 0, 0 };
    v4i_t z4 = { 0, 0, 0, 0 };
#if __clang_major__ >= 23
    v8i_t z8 = { 0, 0, 0, 0, 0, 0, 0, 0 };
    __builtin_amdgcn_tensor_load_to_lds(g0, g1, z4, z4, z8, 0);
#else
    __builtin_amdgcn_tensor_load_to_lds(g0, g1, z4, z4, 0);
#endif
}
#endif

// ============================================================================
// Pre-pass 1: x f32 -> bf16, tiled [m/128][k/32][128][32]
// ============================================================================
__global__ __launch_bounds__(256) void convert_x_bf16(
    const float* __restrict__ x, __bf16* __restrict__ xbf, int total8)
{
    const int t = blockIdx.x * 256 + threadIdx.x;
    if (t >= total8) return;
    const int e = t * 8;
    const int m = e >> 12;          // / IN_F
    const int k = e & (IN_F - 1);
    const float4 v0 = *(const float4*)(x + (size_t)e);
    const float4 v1 = *(const float4*)(x + (size_t)e + 4);
    v8bf o;
    o[0] = (__bf16)v0.x; o[1] = (__bf16)v0.y; o[2] = (__bf16)v0.z; o[3] = (__bf16)v0.w;
    o[4] = (__bf16)v1.x; o[5] = (__bf16)v1.y; o[6] = (__bf16)v1.z; o[7] = (__bf16)v1.w;
    const size_t dst = ((size_t)(m >> 7) * NKT + (k >> 5)) * TILE_ELEMS
                     + (m & 127) * 32 + (k & 31);
    *(v8bf*)(xbf + dst) = o;
}

// ============================================================================
// Pre-pass 2: dequant 4-bit -> bf16 W^T, tiled [n/128][k/32][128 n][32 k]
// ============================================================================
__global__ __launch_bounds__(256) void dequant_w_bf16(
    const int* __restrict__ qweight, const int* __restrict__ qzeros,
    const float* __restrict__ scales, __bf16* __restrict__ wt)
{
    const int t = blockIdx.x * 256 + threadIdx.x;   // word id = kw*OUT_F + n
    const int kw = t >> 12;                         // / OUT_F : 0..511
    const int n  = t & (OUT_F - 1);
    const int k0 = kw * 8;
    const int g  = k0 >> 7;                         // GROUPQ = 128
    const int qw = qweight[t];
    const int zw = qzeros[g * (OUT_F / 8) + (n >> 3)];
    const int z  = (zw >> ((n & 7) * 4)) & 15;
    const float s = scales[g * OUT_F + n];
    v8bf o;
#pragma unroll
    for (int j = 0; j < 8; ++j)
        o[j] = (__bf16)(s * (float)(((qw >> (4 * j)) & 15) - z));
    const size_t dst = ((size_t)(n >> 7) * NKT + (k0 >> 5)) * TILE_ELEMS
                     + (n & 127) * 32 + (k0 & 31);
    *(v8bf*)(wt + dst) = o;
}

// ============================================================================
// Main GEMM: bf16 tiles (L2-resident) -> LDS via TDM (double-buffered) -> WMMA
// ============================================================================
__global__ __launch_bounds__(256) void gemm_bf16_tdm(
    const __bf16* __restrict__ xbf, const __bf16* __restrict__ wt,
    const float* __restrict__ bias, float* __restrict__ out)
{
    __shared__ __bf16 ldsA[2][TILE_LDS];
    __shared__ __bf16 ldsB[2][TILE_LDS];

    const int tid  = threadIdx.x;
    const int lane = tid & 31;
    const int wv   = tid >> 5;
    const int wm   = wv >> 2;           // 0..1
    const int wn   = wv & 3;            // 0..3
    const int halfSel = lane >> 4;
    const int l16     = lane & 15;

    const __bf16* aT = xbf + (size_t)blockIdx.y * NKT * TILE_ELEMS;
    const __bf16* bT = wt  + (size_t)blockIdx.x * NKT * TILE_ELEMS;

    v8f acc[4][2] = {};

    // wave 0 drives both TDM descriptors; DMA runs async under the WMMAs
    auto stage = [&](int buf, int kt) {
#if HAVE_TDM
        if (wv == 0) {
            tdm_tile_to_lds(aT + (size_t)kt * TILE_ELEMS, (unsigned)(size_t)&ldsA[buf][0]);
            tdm_tile_to_lds(bT + (size_t)kt * TILE_ELEMS, (unsigned)(size_t)&ldsB[buf][0]);
        }
#else
#pragma unroll
        for (int p = 0; p < 2; ++p) {
            const int idx = (tid + p * 256) * 8;      // bf16 elems within tile
            const int row = idx >> 5, col = idx & 31;
            const v8bf va = *(const v8bf*)(aT + (size_t)kt * TILE_ELEMS + idx);
            const v8bf vb = *(const v8bf*)(bT + (size_t)kt * TILE_ELEMS + idx);
            *(v8bf*)&ldsA[buf][row * LDT + col] = va;
            *(v8bf*)&ldsB[buf][row * LDT + col] = vb;
        }
#endif
    };
    // SOPP wait is EXEC-independent; waves with TENSORcnt==0 pass for free
    auto stage_wait = [&]() {
#if HAVE_TDM
        __builtin_amdgcn_s_wait_tensorcnt(0);
#endif
        __syncthreads();
    };

    auto compute = [&](int cur) {
        V16U a[4], b[2];
#pragma unroll
        for (int i = 0; i < 4; ++i) {
            const __bf16* src = &ldsA[cur][(wm * 64 + i * 16 + l16) * LDT + halfSel * 8];
            a[i].h[0] = *(const v8bf*)(src);
            a[i].h[1] = *(const v8bf*)(src + 16);
        }
#pragma unroll
        for (int j = 0; j < 2; ++j) {
            const __bf16* src = &ldsB[cur][(wn * 32 + j * 16 + l16) * LDT + halfSel * 8];
            b[j].h[0] = *(const v8bf*)(src);
            b[j].h[1] = *(const v8bf*)(src + 16);
        }
#pragma unroll
        for (int i = 0; i < 4; ++i)
#pragma unroll
            for (int j = 0; j < 2; ++j)
                acc[i][j] = __builtin_amdgcn_wmma_f32_16x16x32_bf16(
                    false, a[i].v, false, b[j].v, (short)0, acc[i][j], false, false);
    };

    // ---- branch-free pipelined main loop (last iteration peeled) ----
    stage(0, 0);
    stage_wait();
    for (int kt = 0; kt < NKT - 1; ++kt) {
        stage((kt + 1) & 1, kt + 1);   // DMA for next tile overlaps compute
        compute(kt & 1);
        stage_wait();
    }
    compute((NKT - 1) & 1);

    // ---- epilogue: fused bias, f32 store ----
#pragma unroll
    for (int j = 0; j < 2; ++j) {
        const int col = blockIdx.x * BN + wn * 32 + j * 16 + l16;
        const float bv = bias[col];
#pragma unroll
        for (int i = 0; i < 4; ++i) {
            const int mbase = blockIdx.y * BM + wm * 64 + i * 16 + halfSel * 8;
#pragma unroll
            for (int r = 0; r < 8; ++r)
                out[(size_t)(mbase + r) * OUT_F + col] = acc[i][j][r] + bv;
        }
    }
}

// ============================================================================
// Fallback: fused kernel (used only if workspace is too small)
// ============================================================================
__global__ __launch_bounds__(256) void gptq_wmma_bf16(
    const float* __restrict__ x, const int* __restrict__ qweight,
    const int* __restrict__ qzeros, const float* __restrict__ scales,
    const float* __restrict__ bias, float* __restrict__ out)
{
    __shared__ __bf16 ldsA[BM * LDT];
    __shared__ __bf16 ldsB[BN * LDT];

    const int tid  = threadIdx.x;
    const int lane = tid & 31;
    const int wv   = tid >> 5;
    const int wm   = wv >> 2;
    const int wn   = wv & 3;
    const int m0 = blockIdx.y * BM;
    const int n0 = blockIdx.x * BN;
    const int halfSel = lane >> 4;
    const int l16     = lane & 15;

    v8f acc[4][2] = {};

    for (int k0 = 0; k0 < IN_F; k0 += BK) {
#pragma unroll
        for (int p = 0; p < 4; ++p) {
            const int idx = (tid + p * 256) * 4;
            const int row = idx >> 5, kc = idx & 31;
            const float4 v = *(const float4*)(x + (size_t)(m0 + row) * IN_F + k0 + kc);
            __bf16* dst = &ldsA[row * LDT + kc];
            dst[0] = (__bf16)v.x; dst[1] = (__bf16)v.y;
            dst[2] = (__bf16)v.z; dst[3] = (__bf16)v.w;
        }
        const int g = k0 >> 7;
#pragma unroll
        for (int p = 0; p < 2; ++p) {
            const int w = tid + p * 256;
            const int kq = w >> 7, nl = w & 127;
            const int n = n0 + nl;
            const int qw = qweight[(size_t)(k0 / 8 + kq) * OUT_F + n];
            const int zw = qzeros[(size_t)g * (OUT_F / 8) + (n >> 3)];
            const int z = (zw >> ((n & 7) * 4)) & 15;
            const float s = scales[(size_t)g * OUT_F + n];
            __bf16* dst = &ldsB[nl * LDT + kq * 8];
#pragma unroll
            for (int jj = 0; jj < 8; ++jj)
                dst[jj] = (__bf16)(s * (float)(((qw >> (jj * 4)) & 15) - z));
        }
        __syncthreads();

        V16U a[4], b[2];
#pragma unroll
        for (int i = 0; i < 4; ++i) {
            const __bf16* src = &ldsA[(wm * 64 + i * 16 + l16) * LDT + halfSel * 8];
            a[i].h[0] = *(const v8bf*)(src);
            a[i].h[1] = *(const v8bf*)(src + 16);
        }
#pragma unroll
        for (int j = 0; j < 2; ++j) {
            const __bf16* src = &ldsB[(wn * 32 + j * 16 + l16) * LDT + halfSel * 8];
            b[j].h[0] = *(const v8bf*)(src);
            b[j].h[1] = *(const v8bf*)(src + 16);
        }
#pragma unroll
        for (int i = 0; i < 4; ++i)
#pragma unroll
            for (int j = 0; j < 2; ++j)
                acc[i][j] = __builtin_amdgcn_wmma_f32_16x16x32_bf16(
                    false, a[i].v, false, b[j].v, (short)0, acc[i][j], false, false);
        __syncthreads();
    }

#pragma unroll
    for (int j = 0; j < 2; ++j) {
        const int col = n0 + wn * 32 + j * 16 + l16;
        const float bv = bias[col];
#pragma unroll
        for (int i = 0; i < 4; ++i) {
            const int mbase = m0 + wm * 64 + i * 16 + halfSel * 8;
#pragma unroll
            for (int r = 0; r < 8; ++r)
                out[(size_t)(mbase + r) * OUT_F + col] = acc[i][j][r] + bv;
        }
    }
}

// ============================================================================
extern "C" void kernel_launch(void* const* d_in, const int* in_sizes, int n_in,
                              void* d_out, int out_size, void* d_ws, size_t ws_size,
                              hipStream_t stream) {
    const float* x  = (const float*)d_in[0];
    const int*   qw = (const int*)d_in[1];
    const int*   qz = (const int*)d_in[2];
    const float* sc = (const float*)d_in[3];
    const float* bs = (const float*)d_in[4];
    float* out = (float*)d_out;

    const int tokens = in_sizes[0] / IN_F;               // 8192
    const size_t needA = (size_t)tokens * IN_F * sizeof(__bf16);  // 64 MB
    const size_t needB = (size_t)IN_F * OUT_F * sizeof(__bf16);   // 32 MB

    dim3 block(256);
    dim3 grid(OUT_F / BN, tokens / BM);                  // (32, 64)

    if (ws_size >= needA + needB && (tokens % BM) == 0) {
        __bf16* xbf = (__bf16*)d_ws;
        __bf16* wtp = (__bf16*)((char*)d_ws + needA);
        const int total8 = tokens * IN_F / 8;
        hipLaunchKernelGGL(convert_x_bf16, dim3(total8 / 256), block, 0, stream,
                           x, xbf, total8);
        hipLaunchKernelGGL(dequant_w_bf16, dim3((IN_F / 8) * OUT_F / 256), block, 0, stream,
                           qw, qz, sc, wtp);
        hipLaunchKernelGGL(gemm_bf16_tdm, grid, block, 0, stream,
                           xbf, wtp, bs, out);
    } else {
        hipLaunchKernelGGL(gptq_wmma_bf16, grid, block, 0, stream,
                           x, qw, qz, sc, bs, out);
    }
}
